// UTDRecursiveNetOrder2_6176162972383
// MI455X (gfx1250) — matrix operand order repaired and checked by reading, not compiled, not run on hardware
//
#include <hip/hip_runtime.h>

// ---------------------------------------------------------------------------
// Fused UTD recursive net for MI455X (gfx1250, wave32, WMMA).
// One wave owns a 16-row batch tile and scans t = 0..1023. Features staged in
// LDS (f32); all weights resident in VGPRs as pre-packed WMMA A-operands.
//
// Row-permutation trick: producer matmuls write logical row 16*hx + 8*T + r
// into D tile T / VGPR r / lane-half hx, so a D-tile pair is *already* in the
// f16 B-operand lane layout (k = 16*hx + e) -> D->B is 8 v_cvt_pk_f16_f32,
// no ds_bpermute / shfl at all.
// ---------------------------------------------------------------------------

typedef __attribute__((ext_vector_type(16))) _Float16 v16h;
typedef __attribute__((ext_vector_type(8)))  float    v8f;

#define NPTS 1024
#define HID  32

__device__ __forceinline__ float sigm(float x) { return 1.0f / (1.0f + __expf(-x)); }
__device__ __forceinline__ float tanh_fast(float x) { return 2.0f * sigm(2.0f * x) - 1.0f; }

// D = A*B + C, f16 operands, f32 accumulate: v_wmma_f32_16x16x32_f16
__device__ __forceinline__ v8f wmma(v16h a, v16h b, v8f c) {
  return __builtin_amdgcn_wmma_f32_16x16x32_f16(false, a, false, b, (short)0, c,
                                                false, false);
}

// Permuted row for output tile T, tile-local M index m (0..15):
//   row = base + 8*T + (m&7) + 16*(m>>3)
__device__ __forceinline__ int prow(int base, int T, int m) {
  return base + 8 * T + (m & 7) + 16 * (m >> 3);
}

// Pack a 16(out)x32(K) slice of row-major W[OUT][IN] into the gfx1250 f16
// A-operand layout with permuted output rows. K packing (contiguous order, same as B here):
//   lane-half kh: K = ko + kh*16 + e (e = 0..15)
template <int IN>
__device__ __forceinline__ v16h load_A_perm(const float* __restrict__ W, int base,
                                            int T, int ko, int lane) {
  const int row = prow(base, T, lane & 15);
  const int kh  = (lane >> 4) & 1;
  v16h a;
#pragma unroll
  for (int e = 0; e < 8; ++e) a[e] = (_Float16)W[row * IN + ko + kh * 16 + e];
#pragma unroll
  for (int e = 0; e < 8; ++e) a[8 + e] = (_Float16)W[row * IN + ko + kh * 16 + 8 + e];
  return a;
}

// Bias broadcast into a permuted C/D tile: VGPR r -> logical row base+16*hx+8*T+r.
__device__ __forceinline__ v8f bias_tile_perm(const float* __restrict__ b, int base,
                                              int T, int lane) {
  const int hx = (lane >> 4) & 1;
  v8f t;
#pragma unroll
  for (int r = 0; r < 8; ++r) t[r] = b[base + 16 * hx + 8 * T + r];
  return t;
}

// Permuted D-tile pair -> f16 B operand (lane-local, no shuffles).
__device__ __forceinline__ v16h dpair_to_B(v8f d0, v8f d1) {
  v16h b;
#pragma unroll
  for (int r = 0; r < 8; ++r) {
    b[r]     = (_Float16)d0[r];
    b[r + 8] = (_Float16)d1[r];
  }
  return b;
}

template <int F>  // input feature dim: 16 (layer 1) or 32 (layer 2)
__global__ __launch_bounds__(64) void fused_layer(
    const float* __restrict__ feat,  // [NPTS, F]
    const float* __restrict__ Wp,    // [HID, 2F]
    const float* __restrict__ bp,    // [HID]
    const float* __restrict__ Wih,   // [3*HID, HID]
    const float* __restrict__ Whh,   // [3*HID, HID]
    const float* __restrict__ bih,   // [3*HID]
    const float* __restrict__ bhh,   // [3*HID]
    float* __restrict__ hout)        // [NPTS, HID]
{
  constexpr int KC = (2 * F) / 32;      // K-chunks in pairwise matmul (1 or 2)
  __shared__ float sfeat[NPTS * F];     // 64KB (F=16) / 128KB (F=32) of 320KB WGP LDS

  const int tid  = threadIdx.x;
  const int lane = tid & 31;
  const int wave = tid >> 5;
  const int hx   = (lane >> 4) & 1;
  const int b    = (blockIdx.x * 2 + wave) * 16 + (lane & 15);  // batch row

  // Stage all features into LDS once (vectorized b128 copies).
  {
    const float4* g4 = reinterpret_cast<const float4*>(feat);
    float4*       s4 = reinterpret_cast<float4*>(sfeat);
    for (int i = tid; i < (NPTS * F) / 4; i += blockDim.x) s4[i] = g4[i];
  }
  __syncthreads();

  // Per-lane f_b slice feeding this lane's diffs B-operand elements.
  const int off0 = (F > 16) ? hx * 16 : 0;
  float fb[16];
#pragma unroll
  for (int e = 0; e < 16; ++e) fb[e] = feat[b * F + off0 + e];
  // F==16 predication folded into an FMA: val = fbs[e] + fsc*ft[e]
  const float fsc = hx ? 1.0f : 0.0f;
  float fbs[16];
#pragma unroll
  for (int e = 0; e < 16; ++e) fbs[e] = hx ? -fb[e] : fb[e];

  // ---- loop-invariant weights as resident, row-permuted WMMA A-operands ----
  v16h Ap[2 * KC];
#pragma unroll
  for (int T = 0; T < 2; ++T)
#pragma unroll
    for (int c = 0; c < KC; ++c) Ap[T * KC + c] = load_A_perm<2 * F>(Wp, 0, T, c * 32, lane);
  v16h Aih[6], Ahh[6];
#pragma unroll
  for (int g = 0; g < 3; ++g)
#pragma unroll
    for (int T = 0; T < 2; ++T) {
      Aih[g * 2 + T] = load_A_perm<HID>(Wih, 32 * g, T, 0, lane);
      Ahh[g * 2 + T] = load_A_perm<HID>(Whh, 32 * g, T, 0, lane);
    }
  v8f cbp[2];
#pragma unroll
  for (int T = 0; T < 2; ++T) cbp[T] = bias_tile_perm(bp, 0, T, lane);
  // r/z gates: only bih+bhh matters (summed); n gate needs them separate.
  v8f cbx[6], cbhn[2];
#pragma unroll
  for (int o = 0; o < 4; ++o) {
    v8f bi = bias_tile_perm(bih, 32 * (o >> 1), o & 1, lane);
    v8f bh = bias_tile_perm(bhh, 32 * (o >> 1), o & 1, lane);
#pragma unroll
    for (int r = 0; r < 8; ++r) cbx[o][r] = bi[r] + bh[r];
  }
#pragma unroll
  for (int T = 0; T < 2; ++T) {
    cbx[4 + T]  = bias_tile_perm(bih, 64, T, lane);
    cbhn[T]     = bias_tile_perm(bhh, 64, T, lane);
  }
  const v8f zc = {};  // zero C for the r/z recurrent WMMAs

  v8f hD[2] = {};  // hidden state 32x16 as two permuted f32 tiles, h0 = 0

  for (int t = 0; t < NPTS; ++t) {
    // f_t slice from LDS: 64B contiguous -> ds_load_b128 x4 (bank broadcast)
    float ft[16];
    {
      const float4* ftp = reinterpret_cast<const float4*>(sfeat + t * F + off0);
#pragma unroll
      for (int q = 0; q < 4; ++q) {
        float4 v = ftp[q];
        ft[4 * q + 0] = v.x; ft[4 * q + 1] = v.y;
        ft[4 * q + 2] = v.z; ft[4 * q + 3] = v.w;
      }
    }

    // diffs B operand: cat([f_b, f_t - f_b]) in natural K order
    v16h dB[KC];
    if constexpr (KC == 1) {
#pragma unroll
      for (int e = 0; e < 16; ++e) dB[0][e] = (_Float16)fmaf(fsc, ft[e], fbs[e]);
    } else {
#pragma unroll
      for (int e = 0; e < 16; ++e) dB[0][e] = (_Float16)fb[e];
#pragma unroll
      for (int e = 0; e < 16; ++e) dB[1][e] = (_Float16)(ft[e] - fb[e]);
    }

    // u = relu(Wp @ diffs + bp), produced directly in permuted-B-ready layout
    v8f u0 = cbp[0], u1 = cbp[1];
#pragma unroll
    for (int c = 0; c < KC; ++c) {
      u0 = wmma(Ap[c],      dB[c], u0);
      u1 = wmma(Ap[KC + c], dB[c], u1);
    }
#pragma unroll
    for (int r = 0; r < 8; ++r) {
      u0[r] = fmaxf(u0[r], 0.0f);
      u1[r] = fmaxf(u1[r], 0.0f);
    }

    v16h uB = dpair_to_B(u0, u1);        // lane-local cvt only
    v16h hB = dpair_to_B(hD[0], hD[1]);

    // gates (PyTorch order r,z,n), tiles element-aligned with hD
#pragma unroll
    for (int tl = 0; tl < 2; ++tl) {
      v8f xr = wmma(Aih[0 + tl], uB, cbx[0 + tl]);
      v8f hr = wmma(Ahh[0 + tl], hB, zc);
      v8f xz = wmma(Aih[2 + tl], uB, cbx[2 + tl]);
      v8f hz = wmma(Ahh[2 + tl], hB, zc);
      v8f xn = wmma(Aih[4 + tl], uB, cbx[4 + tl]);
      v8f hn = wmma(Ahh[4 + tl], hB, cbhn[tl]);
#pragma unroll
      for (int r = 0; r < 8; ++r) {
        float rg  = sigm(xr[r] + hr[r]);
        float zg  = sigm(xz[r] + hz[r]);
        float ng  = tanh_fast(xn[r] + rg * hn[r]);
        hD[tl][r] = (1.0f - zg) * ng + zg * hD[tl][r];
      }
    }
  }

  // Final hidden state: tile tl, VGPR r -> hidden index 16*hx + 8*tl + r
#pragma unroll
  for (int tl = 0; tl < 2; ++tl)
#pragma unroll
    for (int r = 0; r < 8; ++r) hout[b * HID + 16 * hx + 8 * tl + r] = hD[tl][r];
}

// Tiny 32->3 classifier; bandwidth-trivial, plain VALU.
__global__ __launch_bounds__(128) void classifier(const float* __restrict__ h,
                                                  const float* __restrict__ W,
                                                  const float* __restrict__ bias,
                                                  float* __restrict__ out) {
  int i = blockIdx.x * blockDim.x + threadIdx.x;
  if (i >= NPTS) return;
  float a0 = bias[0], a1 = bias[1], a2 = bias[2];
#pragma unroll
  for (int k = 0; k < HID; ++k) {
    float hv = h[i * HID + k];
    a0 = fmaf(hv, W[0 * HID + k], a0);
    a1 = fmaf(hv, W[1 * HID + k], a1);
    a2 = fmaf(hv, W[2 * HID + k], a2);
  }
  out[i * 3 + 0] = a0;
  out[i * 3 + 1] = a1;
  out[i * 3 + 2] = a2;
}

extern "C" void kernel_launch(void* const* d_in, const int* in_sizes, int n_in,
                              void* d_out, int out_size, void* d_ws, size_t ws_size,
                              hipStream_t stream) {
  const float* x     = (const float*)d_in[0];
  const float* p1w   = (const float*)d_in[1];
  const float* p1b   = (const float*)d_in[2];
  const float* g1wih = (const float*)d_in[3];
  const float* g1whh = (const float*)d_in[4];
  const float* g1bih = (const float*)d_in[5];
  const float* g1bhh = (const float*)d_in[6];
  const float* p2w   = (const float*)d_in[7];
  const float* p2b   = (const float*)d_in[8];
  const float* g2wih = (const float*)d_in[9];
  const float* g2whh = (const float*)d_in[10];
  const float* g2bih = (const float*)d_in[11];
  const float* g2bhh = (const float*)d_in[12];
  const float* clfw  = (const float*)d_in[13];
  const float* clfb  = (const float*)d_in[14];

  float* h1 = (float*)d_ws;            // [1024,32]
  float* h2 = h1 + NPTS * HID;         // [1024,32]

  dim3 blk(64);                        // 2 waves * 16 batch rows per block
  dim3 grd(NPTS / 32);                 // 32 blocks -> 64 recurrence waves

  fused_layer<16><<<grd, blk, 0, stream>>>(x,  p1w, p1b, g1wih, g1whh, g1bih, g1bhh, h1);
  fused_layer<32><<<grd, blk, 0, stream>>>(h1, p2w, p2b, g2wih, g2whh, g2bih, g2bhh, h2);
  classifier<<<dim3(NPTS / 128), dim3(128), 0, stream>>>(h2, clfw, clfb, (float*)d_out);
}
